// PhaseAssociativeLayer_82858509074619
// MI455X (gfx1250) — compile-verified, matrix operand order checked by reading, not run on hardware
//
#include <hip/hip_runtime.h>

// ---- problem constants ----
#define B_    2
#define T_    1024
#define DIM_  1024
#define H_    16
#define D_    64
#define INNER_ 1024
#define M_    (B_ * T_)   // 2048

typedef __attribute__((ext_vector_type(16))) __bf16 v16bf;
typedef __attribute__((ext_vector_type(8)))  __bf16 v8bf;
typedef __attribute__((ext_vector_type(8)))  float  v8f;

__device__ inline v8f wmma_bf16(v16bf a, v16bf b, v8f c) {
    return __builtin_amdgcn_wmma_f32_16x16x32_bf16(false, a, false, b, (short)0, c,
                                                   false, false);
}

// Load a 16-element bf16 A/B fragment: two contiguous 8-element runs at p and p+16.
// Element j holds k = (j>>3)*16 + koff + (j&7) relative to (p - koff).
__device__ inline v16bf frag_ld(const __bf16* p) {
    v8bf lo = *(const v8bf*)(p);
    v8bf hi = *(const v8bf*)(p + 16);
    v16bf r;
#pragma unroll
    for (int j = 0; j < 8; ++j) { r[j] = lo[j]; r[j + 8] = hi[j]; }
    return r;
}

// ---------------- split interleaved complex fp32 -> planar bf16 ----------------
__global__ void split_to_bf16(const float* __restrict__ in,
                              __bf16* __restrict__ outr, __bf16* __restrict__ outi,
                              int n) {
    int i = blockIdx.x * blockDim.x + threadIdx.x;
    if (i >= n) return;
    outr[i] = (__bf16)in[2 * i];
    outi[i] = (__bf16)in[2 * i + 1];
}

// ---------------- gates: dt -> softplus, p -> sigmoid, gamma ----------------
__global__ void gate_kernel(const float* __restrict__ x,
                            const float* __restrict__ dt_w, const float* __restrict__ dt_b,
                            const float* __restrict__ dt_bias,
                            const float* __restrict__ pg_w, const float* __restrict__ pg_b,
                            float* __restrict__ p_out, float* __restrict__ g_out) {
    int bt = blockIdx.x;                       // 0 .. B*T-1
    const float* xrow = x + (size_t)bt * DIM_ * 2;
    __shared__ float s_dt[H_];
    __shared__ float s_p[H_];
    if (threadIdx.x < H_) { s_dt[threadIdx.x] = 0.f; s_p[threadIdx.x] = 0.f; }
    __syncthreads();

    float acc_dt[H_], acc_p[H_];
#pragma unroll
    for (int h = 0; h < H_; ++h) { acc_dt[h] = 0.f; acc_p[h] = 0.f; }

    for (int k = threadIdx.x; k < DIM_; k += 256) {
        float vr = xrow[2 * k], vi = xrow[2 * k + 1];
        float mag = sqrtf(vr * vr + vi * vi);
#pragma unroll
        for (int h = 0; h < H_; ++h) {
            acc_dt[h] += vr * dt_w[h * (2 * DIM_) + k] + vi * dt_w[h * (2 * DIM_) + DIM_ + k];
            acc_p[h]  += mag * pg_w[h * DIM_ + k];
        }
    }
#pragma unroll
    for (int h = 0; h < H_; ++h) {
        atomicAdd(&s_dt[h], acc_dt[h]);
        atomicAdd(&s_p[h],  acc_p[h]);
    }
    __syncthreads();
    if (threadIdx.x < H_) {
        int h = threadIdx.x;
        float z  = s_dt[h] + dt_b[h] + dt_bias[h];
        float dt = (z > 20.f) ? z : log1pf(expf(z));
        float pp = 1.f / (1.f + expf(-(s_p[h] + pg_b[h])));
        float gm = expf(-dt) * (1.f - pp) + pp;
        int b = bt / T_, t = bt % T_;
        size_t o = ((size_t)(b * H_ + h)) * T_ + t;
        p_out[o] = pp;
        g_out[o] = gm;
    }
}

// ---------------- cumulative decay C = cumsum(-log(gamma + 1e-8)) ----------------
__global__ void cumsum_kernel(const float* __restrict__ g, float* __restrict__ C) {
    int bh = threadIdx.x;
    if (bh >= B_ * H_) return;
    float c = 0.f;
    for (int t = 0; t < T_; ++t) {
        c += -logf(g[(size_t)bh * T_ + t] + 1e-8f);
        C[(size_t)bh * T_ + t] = c;
    }
}

// ---------------- V' = V * (1 - p) ----------------
__global__ void scale_v_kernel(__bf16* __restrict__ Vr, __bf16* __restrict__ Vi,
                               const float* __restrict__ p, int n) {
    int i = blockIdx.x * blockDim.x + threadIdx.x;
    if (i >= n) return;
    int h = (i >> 6) & (H_ - 1);
    int t = (i >> 10) & (T_ - 1);
    int b = i >> 20;
    float s = 1.f - p[((size_t)(b * H_ + h)) * T_ + t];
    Vr[i] = (__bf16)((float)Vr[i] * s);
    Vi[i] = (__bf16)((float)Vi[i] * s);
}

// ---------------- complex GEMM: Y = X * W^T (complex), bf16 in / f32 acc ----------------
// A: [Mdim,Kdim] planar (Ar,Ai). W: [Ndim,Kdim] planar (Wr,Wi).
// mode 0: write bf16 planar Or/Oi [Mdim,Ndim]; mode 1: write interleaved fp32 Of.
__global__ __launch_bounds__(256) void cgemm_kernel(
    const __bf16* __restrict__ Ar, const __bf16* __restrict__ Ai,
    const __bf16* __restrict__ Wr, const __bf16* __restrict__ Wi,
    int Mdim, int Ndim, int Kdim,
    __bf16* __restrict__ Or, __bf16* __restrict__ Oi,
    float* __restrict__ Of, int mode) {
    int lane = threadIdx.x & 31;
    int wave = threadIdx.x >> 5;
    int m0 = blockIdx.x << 4;
    int n0 = (blockIdx.y << 7) + (wave << 4);
    if (n0 >= Ndim) return;
    int koff  = (lane >> 4) << 3;       // 0 or 8
    int lcol  = lane & 15;
    const __bf16* par = Ar + (size_t)(m0 + lcol) * Kdim + koff;
    const __bf16* pai = Ai + (size_t)(m0 + lcol) * Kdim + koff;
    const __bf16* pwr = Wr + (size_t)(n0 + lcol) * Kdim + koff;
    const __bf16* pwi = Wi + (size_t)(n0 + lcol) * Kdim + koff;

    v8f aA = {};   // sum Xr*Wr
    v8f aB = {};   // sum Xi*Wi
    v8f aC = {};   // sum Xr*Wi + Xi*Wr
    for (int k0 = 0; k0 < Kdim; k0 += 32) {
        v16bf fr = frag_ld(par + k0);
        v16bf fi = frag_ld(pai + k0);
        v16bf gr = frag_ld(pwr + k0);
        v16bf gi = frag_ld(pwi + k0);
        aA = wmma_bf16(fr, gr, aA);
        aB = wmma_bf16(fi, gi, aB);
        aC = wmma_bf16(fr, gi, aC);
        aC = wmma_bf16(fi, gr, aC);
    }
    int rbase = (lane >> 4) << 3;
#pragma unroll
    for (int r = 0; r < 8; ++r) {
        int m = m0 + rbase + r;
        int n = n0 + lcol;
        float yr = aA[r] - aB[r];
        float yi = aC[r];
        if (mode == 0) {
            size_t o = (size_t)m * Ndim + n;
            Or[o] = (__bf16)yr;
            Oi[o] = (__bf16)yi;
        } else {
            size_t o = ((size_t)m * Ndim + n) * 2;
            Of[o]     = yr;
            Of[o + 1] = yi;
        }
    }
}

// ---------------- causal decayed complex attention ----------------
// One wave per 16-row t tile. QK^H via wmma; decay+mask in fp32 on the C-layout
// accumulator; (ar, ai, -ai) staged through LDS to become A fragments for A*V'.
__global__ __launch_bounds__(256) void attn_kernel(
    const __bf16* __restrict__ Qr, const __bf16* __restrict__ Qi,
    const __bf16* __restrict__ Kr, const __bf16* __restrict__ Ki,
    const __bf16* __restrict__ Vr, const __bf16* __restrict__ Vi,
    const float* __restrict__ Cbuf,
    __bf16* __restrict__ Yr, __bf16* __restrict__ Yi) {
    __shared__ __attribute__((aligned(16))) __bf16 smem[8 * 3 * 512];
    int lane = threadIdx.x & 31;
    int wave = threadIdx.x >> 5;
    int b = blockIdx.x >> 4;
    int h = blockIdx.x & (H_ - 1);
    int t0 = (blockIdx.y << 7) + (wave << 4);
    int koff  = (lane >> 4) << 3;      // 0 or 8
    int rbase = koff;                  // M offset of C-layout rows
    int lcol  = lane & 15;
    const float* Crow = Cbuf + (size_t)blockIdx.x * T_;

    __bf16* lar = smem + wave * 1536;  // 16x32 ar tile
    __bf16* lai = lar + 512;           // 16x32 ai tile
    __bf16* lan = lar + 1024;          // 16x32 -ai tile

    // Q A-fragments (row = t, k = d), D = 64 -> two K=32 fragments per component
    const __bf16* qrp = Qr + ((size_t)(b * T_ + t0 + lcol) * H_ + h) * D_ + koff;
    const __bf16* qip = Qi + ((size_t)(b * T_ + t0 + lcol) * H_ + h) * D_ + koff;
    v16bf qr0 = frag_ld(qrp), qr1 = frag_ld(qrp + 32);
    v16bf qi0 = frag_ld(qip), qi1 = frag_ld(qip + 32);

    float Ct[8];
#pragma unroll
    for (int r = 0; r < 8; ++r) Ct[r] = Crow[t0 + rbase + r];

    v8f accR[4] = {{}, {}, {}, {}};
    v8f accI[4] = {{}, {}, {}, {}};

    for (int s0 = 0; s0 < t0 + 16; s0 += 32) {
#pragma unroll
        for (int sub = 0; sub < 2; ++sub) {
            int ss = s0 + (sub << 4);
            int srow = ss + lcol;
            const __bf16* krp = Kr + ((size_t)(b * T_ + srow) * H_ + h) * D_ + koff;
            const __bf16* kip = Ki + ((size_t)(b * T_ + srow) * H_ + h) * D_ + koff;
            v16bf kr0 = frag_ld(krp), kr1 = frag_ld(krp + 32);
            v16bf ki0 = frag_ld(kip), ki1 = frag_ld(kip + 32);
            v8f wA = {};   // Qr.Kr + Qi.Ki
            v8f wB = {};   // Qi.Kr
            v8f wC = {};   // Qr.Ki
            wA = wmma_bf16(qr0, kr0, wA);
            wA = wmma_bf16(qr1, kr1, wA);
            wA = wmma_bf16(qi0, ki0, wA);
            wA = wmma_bf16(qi1, ki1, wA);
            wB = wmma_bf16(qi0, kr0, wB);
            wB = wmma_bf16(qi1, kr1, wB);
            wC = wmma_bf16(qr0, ki0, wC);
            wC = wmma_bf16(qr1, ki1, wC);
            float Cs = Crow[srow];
#pragma unroll
            for (int r = 0; r < 8; ++r) {
                int trow = t0 + rbase + r;
                float dm = (srow <= trow) ? __expf(Cs - Ct[r]) : 0.f;
                float arv = wA[r] * dm;
                float aiv = (wB[r] - wC[r]) * dm;
                int off = (rbase + r) * 32 + (sub << 4) + lcol;
                lar[off] = (__bf16)arv;
                lai[off] = (__bf16)aiv;
                lan[off] = (__bf16)(-aiv);
            }
        }
        // wait for this wave's LDS tile writes before fragment re-load (intra-wave)
        asm volatile("s_wait_dscnt 0" ::: "memory");

        // A fragments of the score tile (row = t-local, k = s-local 0..31)
        const __bf16* arp = lar + lcol * 32 + koff;
        v16bf far = frag_ld(arp);
        v16bf fai = frag_ld(arp + 512);
        v16bf fan = frag_ld(arp + 1024);
#pragma unroll
        for (int n = 0; n < 4; ++n) {
            int dcol = (n << 4) + lcol;
            v16bf bvr, bvi;
#pragma unroll
            for (int j = 0; j < 16; ++j) {
                int sloc = ((j >> 3) << 4) + koff + (j & 7);
                size_t vo = ((size_t)(b * T_ + s0 + sloc) * H_ + h) * D_ + dcol;
                bvr[j] = Vr[vo];
                bvi[j] = Vi[vo];
            }
            accR[n] = wmma_bf16(far, bvr, accR[n]);   // + ar*Vr
            accR[n] = wmma_bf16(fan, bvi, accR[n]);   // - ai*Vi
            accI[n] = wmma_bf16(far, bvi, accI[n]);   // + ar*Vi
            accI[n] = wmma_bf16(fai, bvr, accI[n]);   // + ai*Vr
        }
    }

#pragma unroll
    for (int n = 0; n < 4; ++n) {
#pragma unroll
        for (int r = 0; r < 8; ++r) {
            int trow = t0 + rbase + r;
            int d = (n << 4) + lcol;
            size_t o = ((size_t)(b * T_ + trow) * H_ + h) * D_ + d;
            Yr[o] = (__bf16)accR[n][r];
            Yi[o] = (__bf16)accI[n][r];
        }
    }
}

// ---------------- launch ----------------
extern "C" void kernel_launch(void* const* d_in, const int* in_sizes, int n_in,
                              void* d_out, int out_size, void* d_ws, size_t ws_size,
                              hipStream_t stream) {
    (void)in_sizes; (void)n_in; (void)out_size; (void)ws_size;
    const float* x      = (const float*)d_in[0];
    const float* qw     = (const float*)d_in[1];
    const float* kw     = (const float*)d_in[2];
    const float* vw     = (const float*)d_in[3];
    const float* ow     = (const float*)d_in[4];
    const float* dt_w   = (const float*)d_in[5];
    const float* dt_b   = (const float*)d_in[6];
    const float* dt_bias= (const float*)d_in[7];
    const float* pg_w   = (const float*)d_in[8];
    const float* pg_b   = (const float*)d_in[9];

    char* ws = (char*)d_ws;
    const size_t MB = 1024ull * 1024ull;
    __bf16* Xr  = (__bf16*)(ws + 0 * MB);
    __bf16* Xi  = (__bf16*)(ws + 4 * MB);
    __bf16* QWr = (__bf16*)(ws + 8 * MB);
    __bf16* QWi = (__bf16*)(ws + 10 * MB);
    __bf16* KWr = (__bf16*)(ws + 12 * MB);
    __bf16* KWi = (__bf16*)(ws + 14 * MB);
    __bf16* VWr = (__bf16*)(ws + 16 * MB);
    __bf16* VWi = (__bf16*)(ws + 18 * MB);
    __bf16* OWr = (__bf16*)(ws + 20 * MB);
    __bf16* OWi = (__bf16*)(ws + 22 * MB);
    __bf16* Qrb = (__bf16*)(ws + 24 * MB);
    __bf16* Qib = (__bf16*)(ws + 28 * MB);
    __bf16* Krb = (__bf16*)(ws + 32 * MB);
    __bf16* Kib = (__bf16*)(ws + 36 * MB);
    __bf16* Vrb = (__bf16*)(ws + 40 * MB);
    __bf16* Vib = (__bf16*)(ws + 44 * MB);
    __bf16* Yrb = (__bf16*)(ws + 48 * MB);
    __bf16* Yib = (__bf16*)(ws + 52 * MB);
    float*  Pbuf = (float*)(ws + 56 * MB);
    float*  Gbuf = (float*)(ws + 56 * MB + 128 * 1024);
    float*  Cbuf = (float*)(ws + 56 * MB + 256 * 1024);

    // 1) split fp32 complex -> planar bf16
    split_to_bf16<<<8192, 256, 0, stream>>>(x,  Xr,  Xi,  M_ * DIM_);
    split_to_bf16<<<4096, 256, 0, stream>>>(qw, QWr, QWi, INNER_ * DIM_);
    split_to_bf16<<<4096, 256, 0, stream>>>(kw, KWr, KWi, INNER_ * DIM_);
    split_to_bf16<<<4096, 256, 0, stream>>>(vw, VWr, VWi, INNER_ * DIM_);
    split_to_bf16<<<4096, 256, 0, stream>>>(ow, OWr, OWi, DIM_ * INNER_);

    // 2) gates + cumulative decay
    gate_kernel<<<B_ * T_, 256, 0, stream>>>(x, dt_w, dt_b, dt_bias, pg_w, pg_b, Pbuf, Gbuf);
    cumsum_kernel<<<1, 32, 0, stream>>>(Gbuf, Cbuf);

    // 3) Q/K/V complex projections (WMMA)
    dim3 gg(M_ / 16, INNER_ / 128);
    cgemm_kernel<<<gg, 256, 0, stream>>>(Xr, Xi, QWr, QWi, M_, INNER_, DIM_, Qrb, Qib, nullptr, 0);
    cgemm_kernel<<<gg, 256, 0, stream>>>(Xr, Xi, KWr, KWi, M_, INNER_, DIM_, Krb, Kib, nullptr, 0);
    cgemm_kernel<<<gg, 256, 0, stream>>>(Xr, Xi, VWr, VWi, M_, INNER_, DIM_, Vrb, Vib, nullptr, 0);

    // 4) V' = V * (1-p)
    scale_v_kernel<<<8192, 256, 0, stream>>>(Vrb, Vib, Pbuf, M_ * INNER_);

    // 5) causal decayed complex attention (WMMA)
    attn_kernel<<<dim3(B_ * H_, T_ / 128), 256, 0, stream>>>(Qrb, Qib, Krb, Kib, Vrb, Vib,
                                                             Cbuf, Yrb, Yib);

    // 6) output complex projection -> interleaved fp32 d_out
    dim3 go(M_ / 16, DIM_ / 128);
    cgemm_kernel<<<go, 256, 0, stream>>>(Yrb, Yib, OWr, OWi, M_, DIM_, INNER_,
                                         nullptr, nullptr, (float*)d_out, 1);
}
